// CrossMultiHeadAttention_87316685128327
// MI455X (gfx1250) — matrix operand order repaired and checked by reading, not compiled, not run on hardware
//
#include <hip/hip_runtime.h>
#include <hip/hip_bf16.h>
#include <math.h>

// ---------------- problem constants ----------------
#define EMB   1024
#define HEADS 16
#define DEPTH 64
#define BATCH 2
#define LQ    1024
#define LKV   2048

typedef __attribute__((ext_vector_type(16))) __bf16 v16bf;
typedef __attribute__((ext_vector_type(8)))  float  v8f;
typedef __attribute__((ext_vector_type(4)))  int    v4i;

// ---- CDNA5 async global->LDS path (guarded: falls back to VGPR staging) ----
#if __has_builtin(__builtin_amdgcn_global_load_async_to_lds_b128) && \
    __has_builtin(__builtin_amdgcn_s_wait_asynccnt)
#define HAVE_ASYNC_LDS 1
typedef __attribute__((address_space(1))) v4i* as1_v4i;
typedef __attribute__((address_space(3))) v4i* as3_v4i;
__device__ __forceinline__ void async_copy16(const void* g, void* l) {
    __builtin_amdgcn_global_load_async_to_lds_b128((as1_v4i)g, (as3_v4i)l, 0, 0);
}
#else
#define HAVE_ASYNC_LDS 0
#endif

// Two contiguous 16B LDS loads -> one 16-element bf16 WMMA operand.
__device__ __forceinline__ v16bf ld_pair(const __bf16* p0, const __bf16* p1) {
    union { v16bf v; uint4 u[2]; } r;
    r.u[0] = *(const uint4*)p0;
    r.u[1] = *(const uint4*)p1;
    return r.v;
}

__device__ __forceinline__ v8f wmma_bf16(v16bf a, v16bf b, v8f c) {
    // (neg_a, A, neg_b, B, c_mod, C, reuse_a, reuse_b)
    return __builtin_amdgcn_wmma_f32_16x16x32_bf16(false, a, false, b, (short)0, c,
                                                   false, false);
}

// ---------------- kernel 1: fp32 -> bf16 ----------------
__global__ void cvt_f32_bf16(const float* __restrict__ in, __bf16* __restrict__ out, int n) {
    int i = blockIdx.x * 256 + threadIdx.x;
    if (i < n) out[i] = (__bf16)in[i];
}

// ---------------- kernel 2/4: NT GEMM  out[m,n] = sum_e A[m,e] * W[n,e] ----------------
// mode 0: write bf16 split-head layout [B, H, L, D]  (n -> h = n>>6, d = n&63; m -> b = m/L, l = m%L)
// mode 1: write fp32 [M,N] with bias[n]
__global__ __launch_bounds__(256)
void gemm_nt_bf16(const __bf16* __restrict__ A, const __bf16* __restrict__ W,
                  void* __restrict__ outp, const float* __restrict__ bias,
                  int M, int N, int K, int L, int mode) {
    __shared__ __bf16 sA[2][128 * 32];   // 16 KB (double buffered)
    __shared__ __bf16 sW[2][64 * 32];    //  8 KB

    const int tid    = threadIdx.x;
    const int wave   = tid >> 5;          // 0..7
    const int lane   = tid & 31;
    const int laneHi = lane >> 4;         // 0/1
    const int ln     = lane & 15;

    const int row0 = blockIdx.y * 128;
    const int col0 = blockIdx.x * 64;

    v8f acc[4];
    for (int t = 0; t < 4; ++t)
        for (int r = 0; r < 8; ++r) acc[t][r] = 0.0f;

#if HAVE_ASYNC_LDS
    // Each thread issues exactly 3 async copies per tile (2 for A, 1 for W).
    auto issue_tile = [&](int buf, int kkt) {
        for (int j = 0; j < 2; ++j) {
            int u = j * 256 + tid;
            int r = u >> 2, c = u & 3;
            async_copy16(A + (size_t)(row0 + r) * K + kkt + c * 8, (uint4*)sA[buf] + u);
        }
        int r = tid >> 2, c = tid & 3;
        async_copy16(W + (size_t)(col0 + r) * K + kkt + c * 8, (uint4*)sW[buf] + tid);
    };
    issue_tile(0, 0);   // prologue: tile 0 in flight before the loop
#endif

    // One pipeline step; `last` is a compile-time constant at each call site,
    // so the steady-state loop body has an unconditional issue + wait(3).
    auto step = [&](int kk, bool last) {
        const int cur = (kk >> 5) & 1;
        const int nxt = cur ^ 1;
        __syncthreads();   // B1: all waves done reading buf[nxt] (prev cur)
        if (!last && kk + 64 < K) {
            __builtin_prefetch(A + (size_t)(row0 + (tid & 127)) * K + kk + 64, 0, 0);
            if (tid < 64)
                __builtin_prefetch(W + (size_t)(col0 + tid) * K + kk + 64, 0, 0);
        }
#if HAVE_ASYNC_LDS
        if (!last) {
            issue_tile(nxt, kk + 32);                 // overlap next tile's DMA
            __builtin_amdgcn_s_wait_asynccnt(3);      // drain tile `cur` only
        } else {
            __builtin_amdgcn_s_wait_asynccnt(0);
        }
#else
        for (int j = 0; j < 2; ++j) {
            int u = j * 256 + tid;
            int r = u >> 2, c = u & 3;
            ((uint4*)sA[cur])[u] = *(const uint4*)(A + (size_t)(row0 + r) * K + kk + c * 8);
        }
        {
            int r = tid >> 2, c = tid & 3;
            ((uint4*)sW[cur])[tid] = *(const uint4*)(W + (size_t)(col0 + r) * K + kk + c * 8);
        }
#endif
        __syncthreads();   // B2: buf[cur] fully landed for all waves

        // A operand: M = ln (within wave's 16 rows), K runs {0..7,16..23} (+8 for hi half)
        const __bf16* arow = &sA[cur][(wave * 16 + ln) * 32];
        v16bf a = ld_pair(arow + laneHi * 8, arow + 16 + laneHi * 8);
        for (int t = 0; t < 4; ++t) {
            // B operand: N = t*16 + ln, K runs {0..7,16..23} (+8 for hi half)
            const __bf16* brow = &sW[cur][(t * 16 + ln) * 32];
            v16bf b = ld_pair(brow + laneHi * 8, brow + 16 + laneHi * 8);
            acc[t] = wmma_bf16(a, b, acc[t]);
        }
    };

    for (int kk = 0; kk < K - 32; kk += 32) step(kk, false);
    step(K - 32, true);   // peeled final tile: straight-line wait(0)

    // epilogue: C layout -> memory
    for (int t = 0; t < 4; ++t) {
        for (int r = 0; r < 8; ++r) {
            int row = row0 + wave * 16 + r + laneHi * 8;
            int col = col0 + t * 16 + ln;
            float v = acc[t][r];
            if (mode == 0) {
                __bf16* out = (__bf16*)outp;
                int h = col >> 6, d = col & 63;
                int b = row / L, l = row - b * L;
                out[(((size_t)b * HEADS + h) * (size_t)L + l) * DEPTH + d] = (__bf16)v;
            } else {
                float* out = (float*)outp;
                out[(size_t)row * N + col] = v + bias[col];
            }
        }
    }
}

// ---------------- kernel 3: flash attention ----------------
// grid: (LQ/64, HEADS, BATCH), block: 128 (4 waves). Each wave owns a 16x64 output tile.
__global__ __launch_bounds__(128)
void flash_attn(const __bf16* __restrict__ Qm, const __bf16* __restrict__ Km,
                const __bf16* __restrict__ Vm, const int* __restrict__ mask,
                __bf16* __restrict__ Out) {
    __shared__ __bf16 sQ[64 * 64];       // 8 KB  (64 q-rows x D)
    __shared__ __bf16 sK[2][32 * 64];    // 8 KB  (kv-chunk x D, double buffered)
    __shared__ __bf16 sVt[64 * 32];      // 4 KB  (D x kv-chunk, transposed)
    __shared__ __bf16 sP[4][16 * 32];    // 4 KB  (per-wave P tile, q x kv)

    const int tid    = threadIdx.x;
    const int wave   = tid >> 5;
    const int lane   = tid & 31;
    const int laneHi = lane >> 4;
    const int ln     = lane & 15;

    const int b  = blockIdx.z;
    const int h  = blockIdx.y;
    const int q0 = blockIdx.x * 64;

    const size_t kvbase = ((size_t)b * HEADS + h) * (size_t)LKV * DEPTH;
    const size_t mrow0  = ((size_t)b * LQ + q0 + wave * 16) * (size_t)LKV;

    // load Q block once: 64x64 bf16 = 512 uint4, 4 per thread
    {
        const size_t qbase = (((size_t)b * HEADS + h) * LQ + q0) * DEPTH;
        for (int j = 0; j < 4; ++j) {
            int u = j * 128 + tid;
            ((uint4*)sQ)[u] = ((const uint4*)(Qm + qbase))[u];
        }
    }

#if HAVE_ASYNC_LDS
    // Each thread issues exactly 2 async copies per K chunk.
    auto issue_k = [&](int buf, int kvt) {
        for (int j = 0; j < 2; ++j) {
            int u = j * 128 + tid;
            async_copy16((const uint4*)(Km + kvbase + (size_t)kvt * DEPTH) + u,
                         (uint4*)sK[buf] + u);
        }
    };
    issue_k(0, 0);   // prologue: chunk 0 in flight
#endif

    v8f acc[4];
    for (int t = 0; t < 4; ++t)
        for (int r = 0; r < 8; ++r) acc[t][r] = 0.0f;
    float mrun[8], lrun[8];
    for (int r = 0; r < 8; ++r) { mrun[r] = -3.0e38f; lrun[r] = 0.0f; }

    auto body = [&](int kv, bool last) {
        const int cur = (kv >> 5) & 1;
        const int nxt = cur ^ 1;
        __syncthreads();  // B1: prior readers of sK[nxt]/sVt/sP are done
        if (!last) {
            __builtin_prefetch(Km + kvbase + (size_t)(kv + 32) * DEPTH + tid * 16, 0, 0);
            __builtin_prefetch(Vm + kvbase + (size_t)(kv + 32) * DEPTH + tid * 16, 0, 0);
        }
#if HAVE_ASYNC_LDS
        if (!last) {
            issue_k(nxt, kv + 32);                    // overlap next chunk's DMA
            __builtin_amdgcn_s_wait_asynccnt(2);      // drain chunk `cur` only
        } else {
            __builtin_amdgcn_s_wait_asynccnt(0);
        }
#endif
        // V chunk is transposed into LDS (register path required for transpose)
        for (int j = 0; j < 2; ++j) {
            int u = j * 128 + tid;
#if !HAVE_ASYNC_LDS
            ((uint4*)sK[cur])[u] = ((const uint4*)(Km + kvbase + (size_t)kv * DEPTH))[u];
#endif
            uint4 vv = ((const uint4*)(Vm + kvbase + (size_t)kv * DEPTH))[u];
            int row = u >> 3;          // kv index 0..31
            int cg  = u & 7;           // d-group of 8
            const __bf16* tv = (const __bf16*)&vv;
            for (int jj = 0; jj < 8; ++jj)
                sVt[(cg * 8 + jj) * 32 + row] = tv[jj];
        }
        __syncthreads();  // B2: sK[cur] + sVt landed for all waves

        // ---- scores S(16q x 32kv) = Q . K^T, K-dim = DEPTH = 2 WMMAs per 16-col tile
        v8f s0, s1;
        for (int r = 0; r < 8; ++r) { s0[r] = 0.0f; s1[r] = 0.0f; }
        const __bf16* qrow = &sQ[(wave * 16 + ln) * 64];
        for (int eh = 0; eh < 2; ++eh) {
            v16bf a = ld_pair(qrow + eh * 32 + laneHi * 8,
                              qrow + eh * 32 + 16 + laneHi * 8);
            const __bf16* k0 = &sK[cur][ln * 64 + eh * 32];
            const __bf16* k1 = &sK[cur][(16 + ln) * 64 + eh * 32];
            s0 = wmma_bf16(a, ld_pair(k0 + laneHi * 8, k0 + 16 + laneHi * 8), s0);
            s1 = wmma_bf16(a, ld_pair(k1 + laneHi * 8, k1 + 16 + laneHi * 8), s1);
        }

        // ---- mask + scale + online softmax (rows live in 16-lane halves)
        float alpha[8], p0v[8], p1v[8];
        for (int r = 0; r < 8; ++r) {
            int row = r + laneHi * 8;
            size_t mi = mrow0 + (size_t)row * LKV + kv;
            float v0 = mask[mi + ln]      ? -1.0e9f : s0[r] * 0.125f;
            float v1 = mask[mi + 16 + ln] ? -1.0e9f : s1[r] * 0.125f;
            float mx = fmaxf(v0, v1);
            for (int d = 1; d < 16; d <<= 1) mx = fmaxf(mx, __shfl_xor(mx, d, 16));
            float mnew = fmaxf(mrun[r], mx);
            float al = __expf(mrun[r] - mnew);
            float e0 = __expf(v0 - mnew);
            float e1 = __expf(v1 - mnew);
            float sm = e0 + e1;
            for (int d = 1; d < 16; d <<= 1) sm += __shfl_xor(sm, d, 16);
            lrun[r] = lrun[r] * al + sm;
            mrun[r] = mnew;
            alpha[r] = al;
            p0v[r] = e0; p1v[r] = e1;
        }
        for (int t = 0; t < 4; ++t)
            for (int r = 0; r < 8; ++r) acc[t][r] *= alpha[r];

        // ---- P: C layout -> LDS (q x kv row major) -> A layout
        __bf16* sp = &sP[wave][0];
        for (int r = 0; r < 8; ++r) {
            int row = r + laneHi * 8;
            sp[row * 32 + ln]      = (__bf16)p0v[r];
            sp[row * 32 + 16 + ln] = (__bf16)p1v[r];
        }
        __syncthreads();  // B3: sP visible

        const __bf16* prow = &sp[ln * 32];
        v16bf ap = ld_pair(prow + laneHi * 8, prow + 16 + laneHi * 8);
        for (int t = 0; t < 4; ++t) {
            const __bf16* vrow = &sVt[(t * 16 + ln) * 32];
            v16bf bv = ld_pair(vrow + laneHi * 8, vrow + 16 + laneHi * 8);
            acc[t] = wmma_bf16(ap, bv, acc[t]);
        }
    };

    for (int kv = 0; kv < LKV - 32; kv += 32) body(kv, false);
    body(LKV - 32, true);   // peeled final chunk

    // ---- normalize + write attn output, merged-head layout [B, LQ, E]
    const size_t obase = ((size_t)b * LQ + q0 + wave * 16) * EMB + (size_t)h * DEPTH;
    for (int t = 0; t < 4; ++t) {
        for (int r = 0; r < 8; ++r) {
            int row = r + laneHi * 8;
            int col = t * 16 + ln;
            Out[obase + (size_t)row * EMB + col] = (__bf16)(acc[t][r] / lrun[r]);
        }
    }
}

// ---------------- launch ----------------
extern "C" void kernel_launch(void* const* d_in, const int* in_sizes, int n_in,
                              void* d_out, int out_size, void* d_ws, size_t ws_size,
                              hipStream_t stream) {
    const float* x    = (const float*)d_in[0];
    const float* ctx  = (const float*)d_in[1];
    const int*   mask = (const int*)d_in[2];     // bernoulli bool -> integer
    const float* Wq   = (const float*)d_in[3];
    const float* Wk   = (const float*)d_in[4];
    const float* Wv   = (const float*)d_in[5];
    const float* fcw  = (const float*)d_in[6];
    const float* fcb  = (const float*)d_in[7];
    float* out = (float*)d_out;

    __bf16* ws = (__bf16*)d_ws;
    const size_t M1 = 1024u * 1024u;
    __bf16* xb  = ws;              // 2M
    __bf16* cb  = ws + 2 * M1;     // 4M
    __bf16* wqb = ws + 6 * M1;     // 1M
    __bf16* wkb = ws + 7 * M1;     // 1M
    __bf16* wvb = ws + 8 * M1;     // 1M
    __bf16* fwb = ws + 9 * M1;     // 1M
    __bf16* Qb  = ws + 10 * M1;    // 2M  [B,H,LQ,D]
    __bf16* Kb  = ws + 12 * M1;    // 4M  [B,H,LKV,D]
    __bf16* Vb  = ws + 16 * M1;    // 4M
    __bf16* Ab  = ws + 20 * M1;    // 2M  [B,LQ,E]

    // 1) convert to bf16
    cvt_f32_bf16<<<(int)(2 * M1 / 256), 256, 0, stream>>>(x,   xb,  (int)(2 * M1));
    cvt_f32_bf16<<<(int)(4 * M1 / 256), 256, 0, stream>>>(ctx, cb,  (int)(4 * M1));
    cvt_f32_bf16<<<(int)(M1 / 256),     256, 0, stream>>>(Wq,  wqb, (int)M1);
    cvt_f32_bf16<<<(int)(M1 / 256),     256, 0, stream>>>(Wk,  wkb, (int)M1);
    cvt_f32_bf16<<<(int)(M1 / 256),     256, 0, stream>>>(Wv,  wvb, (int)M1);
    cvt_f32_bf16<<<(int)(M1 / 256),     256, 0, stream>>>(fcw, fwb, (int)M1);

    // 2) projections (NT GEMM, split-head bf16 output)
    gemm_nt_bf16<<<dim3(EMB / 64, (BATCH * LQ) / 128),  256, 0, stream>>>(
        xb, wqb, Qb, nullptr, BATCH * LQ,  EMB, EMB, LQ,  0);
    gemm_nt_bf16<<<dim3(EMB / 64, (BATCH * LKV) / 128), 256, 0, stream>>>(
        cb, wkb, Kb, nullptr, BATCH * LKV, EMB, EMB, LKV, 0);
    gemm_nt_bf16<<<dim3(EMB / 64, (BATCH * LKV) / 128), 256, 0, stream>>>(
        cb, wvb, Vb, nullptr, BATCH * LKV, EMB, EMB, LKV, 0);

    // 3) flash attention
    flash_attn<<<dim3(LQ / 64, HEADS, BATCH), 128, 0, stream>>>(Qb, Kb, Vb, mask, Ab);

    // 4) output projection + bias (fp32)
    gemm_nt_bf16<<<dim3(EMB / 64, (BATCH * LQ) / 128), 256, 0, stream>>>(
        Ab, fwb, (void*)out, fcb, BATCH * LQ, EMB, EMB, 0, 1);
}
